// Multihead_Attention_61581241090547
// MI455X (gfx1250) — compile-verified, matrix operand order checked by reading, not run on hardware
//
#include <hip/hip_runtime.h>
#include <hip/hip_bf16.h>

typedef __attribute__((ext_vector_type(16))) __bf16 bf16x16;
typedef __attribute__((ext_vector_type(8)))  __bf16 bf16x8;
typedef __attribute__((ext_vector_type(4)))  float  f32x4;
typedef __attribute__((ext_vector_type(8)))  float  v8f;

#define NEG_INF_F (-4294967295.0f)   // -2^32 + 1

static __device__ __forceinline__ v8f wmma_bf16(bf16x16 a, bf16x16 b, v8f c) {
  return __builtin_amdgcn_wmma_f32_16x16x32_bf16(false, a, false, b, (short)0, c, false, false);
}

static __device__ __forceinline__ bf16x16 cat8(bf16x8 lo, bf16x8 hi) {
  return __builtin_shufflevector(lo, hi, 0,1,2,3,4,5,6,7,8,9,10,11,12,13,14,15);
}

// Load a 16x32 bf16 A/B fragment from a row-major [row][k] matrix tile.
// Lane l: row = l&15; chunks at k = (l>>4)*8 and 16 + (l>>4)*8 (each 8 bf16 = 16B).
static __device__ __forceinline__ bf16x16 load_frag_bf16(const __bf16* p, int ld) {
  int lane = threadIdx.x & 31;
  const __bf16* r = p + (size_t)(lane & 15) * ld + (lane >> 4) * 8;
  bf16x8 c0 = *(const bf16x8*)(r);
  bf16x8 c1 = *(const bf16x8*)(r + 16);
  return cat8(c0, c1);
}

// Same fragment, but source is f32; convert to bf16 in-register.
static __device__ __forceinline__ bf16x16 load_frag_f32(const float* p, int ld) {
  int lane = threadIdx.x & 31;
  const float* r = p + (size_t)(lane & 15) * ld + (lane >> 4) * 8;
  f32x4 a0 = *(const f32x4*)(r);
  f32x4 a1 = *(const f32x4*)(r + 4);
  f32x4 a2 = *(const f32x4*)(r + 16);
  f32x4 a3 = *(const f32x4*)(r + 20);
  bf16x16 v;
#pragma unroll
  for (int i = 0; i < 4; ++i) {
    v[i]      = (__bf16)a0[i];
    v[4 + i]  = (__bf16)a1[i];
    v[8 + i]  = (__bf16)a2[i];
    v[12 + i] = (__bf16)a3[i];
  }
  return v;
}

// ---------------- f32 -> bf16 bulk convert ----------------
__global__ __launch_bounds__(256) void cvt_bf16_kernel(const float* __restrict__ src,
                                                       unsigned short* __restrict__ dst,
                                                       int nelem) {
  int i = (blockIdx.x * 256 + threadIdx.x) * 4;
  if (i + 3 < nelem) {
    f32x4 v = *(const f32x4*)(src + i);
    __bf16* d = (__bf16*)dst;
#pragma unroll
    for (int j = 0; j < 4; ++j) d[i + j] = (__bf16)v[j];
  }
}

// ---------------- padding masks: sign(sum|row|) ----------------
__global__ __launch_bounds__(256) void mask_kernel(const float* __restrict__ X,
                                                   float* __restrict__ mask) {
  int lane = threadIdx.x & 31;
  int row = blockIdx.x * 8 + (threadIdx.x >> 5);   // 8192 rows total
  const float* r = X + (size_t)row * 1024;
  float s = 0.0f;
#pragma unroll 4
  for (int i = lane; i < 1024; i += 32) s += fabsf(r[i]);
#pragma unroll
  for (int off = 1; off < 32; off <<= 1) s += __shfl_xor(s, off);
  if (lane == 0) mask[row] = (s != 0.0f) ? 1.0f : 0.0f;
}

// ---------------- projection: P = relu(X @ W^T + b), bf16 out ----------------
__global__ __launch_bounds__(256) void proj_kernel(const float* __restrict__ X,
                                                   const unsigned short* __restrict__ W_u,
                                                   const float* __restrict__ bias,
                                                   unsigned short* __restrict__ P_u) {
  const __bf16* W = (const __bf16*)W_u;
  __bf16* P = (__bf16*)P_u;
  int wid = threadIdx.x >> 5;
  int lane = threadIdx.x & 31;
  int h0 = blockIdx.x * 64;
  int m0 = blockIdx.y * 128;
  int wm = (wid & 3) * 32;
  int wh = (wid >> 2) * 32;

  v8f acc[2][2];
#pragma unroll
  for (int i = 0; i < 2; ++i)
#pragma unroll
    for (int j = 0; j < 2; ++j) acc[i][j] = (v8f){0,0,0,0,0,0,0,0};

  const float*  Xb = X + (size_t)(m0 + wm) * 1024;
  const __bf16* Wb = W + (size_t)(h0 + wh) * 1024;

  for (int ks = 0; ks < 32; ++ks) {
    bf16x16 a0 = load_frag_f32(Xb + ks * 32, 1024);
    bf16x16 a1 = load_frag_f32(Xb + (size_t)16 * 1024 + ks * 32, 1024);
    bf16x16 b0 = load_frag_bf16(Wb + ks * 32, 1024);
    bf16x16 b1 = load_frag_bf16(Wb + (size_t)16 * 1024 + ks * 32, 1024);
    acc[0][0] = wmma_bf16(a0, b0, acc[0][0]);
    acc[0][1] = wmma_bf16(a0, b1, acc[0][1]);
    acc[1][0] = wmma_bf16(a1, b0, acc[1][0]);
    acc[1][1] = wmma_bf16(a1, b1, acc[1][1]);
  }

  int col = lane & 15, lh = lane >> 4;
#pragma unroll
  for (int mt = 0; mt < 2; ++mt)
#pragma unroll
    for (int ht = 0; ht < 2; ++ht) {
      int gcol = h0 + wh + ht * 16 + col;
      float bv = bias[gcol];
#pragma unroll
      for (int r = 0; r < 8; ++r) {
        int gm = m0 + wm + mt * 16 + lh * 8 + r;
        float v = acc[mt][ht][r] + bv;
        v = v > 0.0f ? v : 0.0f;
        P[(size_t)gm * 1024 + gcol] = (__bf16)v;
      }
    }
}

// ---------------- attention scores + masked softmax ----------------
// Writes f32 attn_w output; optionally also bf16 probs [h][n][q][t] for the PV GEMM.
// grid: (Tq/32, N, h); block 512 (16 waves, each owns a 64-wide t slice)
__global__ __launch_bounds__(512) void attn_kernel(const unsigned short* __restrict__ Ql_u,
                                                   const unsigned short* __restrict__ Kl_u,
                                                   const float* __restrict__ kmask,
                                                   const float* __restrict__ qmask,
                                                   float* __restrict__ attnw,
                                                   unsigned short* __restrict__ attnb_u) {
  const __bf16* Ql = (const __bf16*)Ql_u;
  const __bf16* Kl = (const __bf16*)Kl_u;
  __bf16* attnb = (__bf16*)attnb_u;
  __shared__ int   s_max[32];
  __shared__ float s_sum[32];

  int tid = threadIdx.x;
  int lane = tid & 31, wid = tid >> 5;
  int q0 = blockIdx.x * 32;
  int n  = blockIdx.y;
  int hh = blockIdx.z;
  int wt0 = wid * 64;

  if (tid < 32) { s_max[tid] = (int)0x80000000; s_sum[tid] = 0.0f; }
  __syncthreads();

  v8f acc[2][4];
#pragma unroll
  for (int i = 0; i < 2; ++i)
#pragma unroll
    for (int j = 0; j < 4; ++j) acc[i][j] = (v8f){0,0,0,0,0,0,0,0};

  const __bf16* Qbase = Ql + (size_t)(n * 1024 + q0) * 1024 + hh * 128;
  const __bf16* Kbase = Kl + (size_t)(n * 1024 + wt0) * 1024 + hh * 128;

#pragma unroll
  for (int ks = 0; ks < 4; ++ks) {
    bf16x16 a0 = load_frag_bf16(Qbase + ks * 32, 1024);
    bf16x16 a1 = load_frag_bf16(Qbase + (size_t)16 * 1024 + ks * 32, 1024);
#pragma unroll
    for (int ct = 0; ct < 4; ++ct) {
      bf16x16 b = load_frag_bf16(Kbase + (size_t)(ct * 16) * 1024 + ks * 32, 1024);
      acc[0][ct] = wmma_bf16(a0, b, acc[0][ct]);
      acc[1][ct] = wmma_bf16(a1, b, acc[1][ct]);
    }
  }

  int col = lane & 15, lh = lane >> 4;
  const float scale = 0.08838834764831843f;  // 1/sqrt(128)

  float km[4];
#pragma unroll
  for (int ct = 0; ct < 4; ++ct) km[ct] = kmask[n * 1024 + wt0 + ct * 16 + col];

#pragma unroll
  for (int qt = 0; qt < 2; ++qt)
#pragma unroll
    for (int ct = 0; ct < 4; ++ct)
#pragma unroll
      for (int r = 0; r < 8; ++r) {
        float v = acc[qt][ct][r] * scale;
        acc[qt][ct][r] = (km[ct] != 0.0f) ? v : NEG_INF_F;
      }

  // row max: values are >= 0 or exactly NEG_INF, so int-compare atomicMax is order-correct
#pragma unroll
  for (int qt = 0; qt < 2; ++qt)
#pragma unroll
    for (int r = 0; r < 8; ++r) {
      float m = fmaxf(fmaxf(acc[qt][0][r], acc[qt][1][r]),
                      fmaxf(acc[qt][2][r], acc[qt][3][r]));
#pragma unroll
      for (int off = 1; off < 16; off <<= 1) m = fmaxf(m, __shfl_xor(m, off));
      if (col == 0) atomicMax(&s_max[qt * 16 + lh * 8 + r], __float_as_int(m));
    }
  __syncthreads();

#pragma unroll
  for (int qt = 0; qt < 2; ++qt)
#pragma unroll
    for (int r = 0; r < 8; ++r) {
      float rm = __int_as_float(s_max[qt * 16 + lh * 8 + r]);
      float s = 0.0f;
#pragma unroll
      for (int ct = 0; ct < 4; ++ct) {
        float p = __expf(acc[qt][ct][r] - rm);
        acc[qt][ct][r] = p;
        s += p;
      }
#pragma unroll
      for (int off = 1; off < 16; off <<= 1) s += __shfl_xor(s, off);
      if (col == 0) atomicAdd(&s_sum[qt * 16 + lh * 8 + r], s);
    }
  __syncthreads();

#pragma unroll
  for (int qt = 0; qt < 2; ++qt)
#pragma unroll
    for (int r = 0; r < 8; ++r) {
      int row = q0 + qt * 16 + lh * 8 + r;
      float inv = qmask[n * 1024 + row] / s_sum[qt * 16 + lh * 8 + r];
      size_t basef = (size_t)(n * 1024 + row) * 8192 + (size_t)hh * 1024 + wt0;
      size_t baseb = (size_t)((hh * 8 + n) * 1024 + row) * 1024 + wt0;
#pragma unroll
      for (int ct = 0; ct < 4; ++ct) {
        float p = acc[qt][ct][r] * inv;
        attnw[basef + ct * 16 + col] = p;
        if (attnb) attnb[baseb + ct * 16 + col] = (__bf16)p;
      }
    }
}

// ---------------- out = attn @ V, all-bf16 operands ----------------
// attnb: bf16 [h][n][q][t]; Vb: bf16 [n][t][c]
// grid: (Cv/128, Tq/128, h*N); block 256 (8 waves, 4x2, wave tile 32q x 64c)
// Double-buffered transposed V tile in LDS: one barrier per k-step.
__global__ __launch_bounds__(256) void pv_bf16_kernel(const unsigned short* __restrict__ attnb_u,
                                                      const unsigned short* __restrict__ Vb_u,
                                                      float* __restrict__ out) {
  const __bf16* attnb = (const __bf16*)attnb_u;
  const __bf16* Vb = (const __bf16*)Vb_u;
  __shared__ __bf16 Vt[2][128 * 32];  // [buf][c][t]
  int tid = threadIdx.x;
  int lane = tid & 31, wid = tid >> 5;
  int c0 = blockIdx.x * 128;
  int q0 = blockIdx.y * 128;
  int hn = blockIdx.z;
  int hh = hn >> 3, n = hn & 7;
  int wm = (wid & 3) * 32;
  int wc = (wid >> 2) * 64;

  v8f acc[2][4];
#pragma unroll
  for (int i = 0; i < 2; ++i)
#pragma unroll
    for (int j = 0; j < 4; ++j) acc[i][j] = (v8f){0,0,0,0,0,0,0,0};

  const __bf16* Abase = attnb + (size_t)((hh * 8 + n) * 1024 + q0 + wm) * 1024;

  int st = tid >> 3;            // staging: row t within tile (0..31)
  int cg = (tid & 7) * 16;      // staging: 16-wide c group (0..112)
  const __bf16* vbase = Vb + (size_t)(n * 1024 + st) * 1024 + c0 + cg;

  // stage tile 0
  {
    bf16x8 x0 = *(const bf16x8*)(vbase);
    bf16x8 x1 = *(const bf16x8*)(vbase + 8);
#pragma unroll
    for (int j = 0; j < 8; ++j) {
      Vt[0][(cg + j) * 32 + st]     = x0[j];
      Vt[0][(cg + 8 + j) * 32 + st] = x1[j];
    }
  }
  __syncthreads();

  for (int ks = 0; ks < 32; ++ks) {
    int cur = ks & 1;
    if (ks < 31) {                       // stage tile ks+1 into the other buffer
      const __bf16* vp = vbase + (size_t)((ks + 1) * 32) * 1024;
      bf16x8 x0 = *(const bf16x8*)(vp);
      bf16x8 x1 = *(const bf16x8*)(vp + 8);
      if (ks < 30) __builtin_prefetch(vp + (size_t)32 * 1024, 0, 0);
      int nxt = cur ^ 1;
#pragma unroll
      for (int j = 0; j < 8; ++j) {
        Vt[nxt][(cg + j) * 32 + st]     = x0[j];
        Vt[nxt][(cg + 8 + j) * 32 + st] = x1[j];
      }
    }

    bf16x16 a0 = load_frag_bf16(Abase + ks * 32, 1024);
    bf16x16 a1 = load_frag_bf16(Abase + (size_t)16 * 1024 + ks * 32, 1024);
#pragma unroll
    for (int ct = 0; ct < 4; ++ct) {
      bf16x16 b = load_frag_bf16(&Vt[cur][(size_t)(wc + ct * 16) * 32], 32);
      acc[0][ct] = wmma_bf16(a0, b, acc[0][ct]);
      acc[1][ct] = wmma_bf16(a1, b, acc[1][ct]);
    }
    __syncthreads();
  }

  int col = lane & 15, lh = lane >> 4;
#pragma unroll
  for (int qt = 0; qt < 2; ++qt)
#pragma unroll
    for (int ct = 0; ct < 4; ++ct)
#pragma unroll
      for (int r = 0; r < 8; ++r) {
        int gq = q0 + wm + qt * 16 + lh * 8 + r;
        int gc = c0 + wc + ct * 16 + col;
        out[(size_t)(n * 1024 + gq) * 8192 + (size_t)hh * 1024 + gc] = acc[qt][ct][r];
      }
}

// ---------------- fallback: out = attn(f32) @ V(f32), convert in-flight ----------------
// grid: (Cv/64, Tq/128, h*N); block 256
__global__ __launch_bounds__(256) void pv_f32_kernel(const float* __restrict__ attnw,
                                                     const float* __restrict__ V,
                                                     float* __restrict__ out) {
  __shared__ __bf16 Vt[64 * 32];
  int tid = threadIdx.x;
  int lane = tid & 31, wid = tid >> 5;
  int c0 = blockIdx.x * 64;
  int q0 = blockIdx.y * 128;
  int hn = blockIdx.z;
  int hh = hn >> 3, n = hn & 7;
  int wm = (wid & 3) * 32;
  int wc = (wid >> 2) * 32;

  v8f acc[2][2];
#pragma unroll
  for (int i = 0; i < 2; ++i)
#pragma unroll
    for (int j = 0; j < 2; ++j) acc[i][j] = (v8f){0,0,0,0,0,0,0,0};

  const float* Abase = attnw + (size_t)(n * 1024 + q0 + wm) * 8192 + (size_t)hh * 1024;

  int st = tid >> 3;
  int cg = (tid & 7) * 8;

  for (int ks = 0; ks < 32; ++ks) {
    const float* vp = V + (size_t)(n * 1024 + ks * 32 + st) * 1024 + c0 + cg;
    f32x4 x0 = *(const f32x4*)vp;
    f32x4 x1 = *(const f32x4*)(vp + 4);
#pragma unroll
    for (int j = 0; j < 4; ++j) {
      Vt[(cg + j) * 32 + st]     = (__bf16)x0[j];
      Vt[(cg + 4 + j) * 32 + st] = (__bf16)x1[j];
    }
    __syncthreads();

    bf16x16 a0 = load_frag_f32(Abase + ks * 32, 8192);
    bf16x16 a1 = load_frag_f32(Abase + (size_t)16 * 8192 + ks * 32, 8192);
    bf16x16 b0 = load_frag_bf16(&Vt[(size_t)wc * 32], 32);
    bf16x16 b1 = load_frag_bf16(&Vt[(size_t)(wc + 16) * 32], 32);
    acc[0][0] = wmma_bf16(a0, b0, acc[0][0]);
    acc[0][1] = wmma_bf16(a0, b1, acc[0][1]);
    acc[1][0] = wmma_bf16(a1, b0, acc[1][0]);
    acc[1][1] = wmma_bf16(a1, b1, acc[1][1]);
    __syncthreads();
  }

  int col = lane & 15, lh = lane >> 4;
#pragma unroll
  for (int qt = 0; qt < 2; ++qt)
#pragma unroll
    for (int ct = 0; ct < 2; ++ct)
#pragma unroll
      for (int r = 0; r < 8; ++r) {
        int gq = q0 + wm + qt * 16 + lh * 8 + r;
        int gc = c0 + wc + ct * 16 + col;
        out[(size_t)(n * 1024 + gq) * 8192 + (size_t)hh * 1024 + gc] = acc[qt][ct][r];
      }
}

extern "C" void kernel_launch(void* const* d_in, const int* in_sizes, int n_in,
                              void* d_out, int out_size, void* d_ws, size_t ws_size,
                              hipStream_t stream) {
  (void)in_sizes; (void)n_in; (void)out_size;
  const float* Q  = (const float*)d_in[0];
  const float* K  = (const float*)d_in[1];
  const float* V  = (const float*)d_in[2];
  const float* Wq = (const float*)d_in[3];
  const float* bq = (const float*)d_in[4];
  const float* Wk = (const float*)d_in[5];
  const float* bk = (const float*)d_in[6];

  float* out   = (float*)d_out;
  float* attnw = out + (size_t)8 * 1024 * 8192;   // second output region

  char* w = (char*)d_ws;
  unsigned short* Wqb = (unsigned short*)w; w += (size_t)1024 * 1024 * 2;
  unsigned short* Wkb = (unsigned short*)w; w += (size_t)1024 * 1024 * 2;
  unsigned short* Qlb = (unsigned short*)w; w += (size_t)8192 * 1024 * 2;
  unsigned short* Klb = (unsigned short*)w; w += (size_t)8192 * 1024 * 2;
  float* qmask = (float*)w; w += (size_t)8192 * 4;
  float* kmask = (float*)w; w += (size_t)8192 * 4;
  unsigned short* Vb    = (unsigned short*)w; w += (size_t)8 * 1024 * 1024 * 2;
  unsigned short* attnb = (unsigned short*)w; w += (size_t)64 * 1024 * 1024 * 2;
  const size_t need_full = (size_t)(w - (char*)d_ws);
  const bool full = ws_size >= need_full;   // ws_size is fixed -> same path every call

  cvt_bf16_kernel<<<1024, 256, 0, stream>>>(Wq, Wqb, 1024 * 1024);
  cvt_bf16_kernel<<<1024, 256, 0, stream>>>(Wk, Wkb, 1024 * 1024);
  if (full) cvt_bf16_kernel<<<8192, 256, 0, stream>>>(V, Vb, 8 * 1024 * 1024);
  mask_kernel<<<1024, 256, 0, stream>>>(Q, qmask);
  mask_kernel<<<1024, 256, 0, stream>>>(K, kmask);
  proj_kernel<<<dim3(16, 64), 256, 0, stream>>>(Q, Wqb, bq, Qlb);
  proj_kernel<<<dim3(16, 64), 256, 0, stream>>>(K, Wkb, bk, Klb);
  attn_kernel<<<dim3(32, 8, 8), 512, 0, stream>>>(Qlb, Klb, kmask, qmask, attnw,
                                                  full ? attnb : (unsigned short*)nullptr);
  if (full)
    pv_bf16_kernel<<<dim3(8, 8, 64), 256, 0, stream>>>(attnb, Vb, out);
  else
    pv_f32_kernel<<<dim3(16, 8, 64), 256, 0, stream>>>(attnw, V, out);
}
